// DistractorScorer_24927990186048
// MI455X (gfx1250) — matrix operand order repaired and checked by reading, not compiled, not run on hardware
//
#include <hip/hip_runtime.h>

// ---------------------------------------------------------------------------
// DistractorScorer for MI455X (gfx1250): bf16 WMMA (v_wmma_f32_16x16x32_bf16)
// for all GEMMs, f32 accumulate, f32 softmax/log-softmax epilogues.
// Register-blocked waves: tproj 3x2 tiles, inners 3x1 tiles (single pass over
// the 151MB att_distr stream), MLP 2x2 tiles.
// ---------------------------------------------------------------------------

#define BN   16      // batch
#define NN   32      // distractors
#define XX   36      // att positions (= Y)
#define XP   48      // padded to multiple of 16
#define DD   2048    // feature dim
#define HH   512     // hidden
#define TWOD 4096
#define RR   512     // BN * NN
#define BIGNEG (-1000000000.0f)

typedef __attribute__((ext_vector_type(16))) __bf16 v16bf;
typedef __attribute__((ext_vector_type(16))) short  v16s;
typedef __attribute__((ext_vector_type(8)))  float  v8f;

union FragU { v16s s; v16bf b; uint4 q[2]; };

__device__ __forceinline__ unsigned short f2bf(float f) {
  unsigned int u = __float_as_uint(f);
  u += 0x7FFFu + ((u >> 16) & 1u);           // round-to-nearest-even
  return (unsigned short)(u >> 16);
}

// A fragment: 16x32 bf16 tile from row-major matrix (K contiguous in memory).
// Contiguous-k layout used here (same in B loaders): lane = m + 16*h; K = 16h + idx (idx 0..15)
__device__ __forceinline__ v16bf frag_a(const unsigned short* src, int ld, int row0, int k0) {
  int lane = threadIdx.x & 31;
  int m = lane & 15, h = lane >> 4;
  const unsigned short* p = src + (size_t)(row0 + m) * ld + k0 + 16 * h;
  FragU f;
  f.q[0] = *(const uint4*)(p);        // idx 0..7
  f.q[1] = *(const uint4*)(p + 8);    // idx 8..15
  return f.b;
}

// B fragment: 32x16 bf16 tile; source stored K-contiguous per column:
// element (k, n) at src[(col0+n)*ld + k].  lane = n + 16*h; K = 16h + idx.
__device__ __forceinline__ v16bf frag_b_kc(const unsigned short* src, int ld, int col0, int k0) {
  int lane = threadIdx.x & 31;
  int n = lane & 15, h = lane >> 4;
  const unsigned short* p = src + (size_t)(col0 + n) * ld + k0 + 16 * h;
  FragU f;
  f.q[0] = *(const uint4*)(p);
  f.q[1] = *(const uint4*)(p + 8);
  return f.b;
}

// B fragment from f32 source (K contiguous per column) with column guard;
// converts f32 -> bf16 on the fly.
__device__ __forceinline__ v16bf frag_b_f32(const float* src, int ld, int col0, int k0, int nvalid) {
  int lane = threadIdx.x & 31;
  int n = lane & 15, h = lane >> 4;
  int col = col0 + n;
  FragU f;
  if (col < nvalid) {
    const float* p = src + (size_t)col * ld + k0 + 16 * h;
    float4 a0 = *(const float4*)(p + 0);
    float4 a1 = *(const float4*)(p + 4);
    float4 a2 = *(const float4*)(p + 8);
    float4 a3 = *(const float4*)(p + 12);
    f.s[0]  = (short)f2bf(a0.x); f.s[1]  = (short)f2bf(a0.y);
    f.s[2]  = (short)f2bf(a0.z); f.s[3]  = (short)f2bf(a0.w);
    f.s[4]  = (short)f2bf(a1.x); f.s[5]  = (short)f2bf(a1.y);
    f.s[6]  = (short)f2bf(a1.z); f.s[7]  = (short)f2bf(a1.w);
    f.s[8]  = (short)f2bf(a2.x); f.s[9]  = (short)f2bf(a2.y);
    f.s[10] = (short)f2bf(a2.z); f.s[11] = (short)f2bf(a2.w);
    f.s[12] = (short)f2bf(a3.x); f.s[13] = (short)f2bf(a3.y);
    f.s[14] = (short)f2bf(a3.z); f.s[15] = (short)f2bf(a3.w);
  } else {
    for (int i = 0; i < 16; ++i) f.s[i] = 0;
  }
  return f.b;
}

__device__ __forceinline__ v8f wmma_bf16(v16bf a, v16bf b, v8f c) {
  return __builtin_amdgcn_wmma_f32_16x16x32_bf16(false, a, false, b, (short)0, c, false, false);
}

// store one 16x16 f32 accumulator tile as bf16 into row-major [*, ld]
__device__ __forceinline__ void store_tile_bf16(unsigned short* dst, int ld,
                                                int row0, int col0, v8f c) {
  int lane = threadIdx.x & 31;
  int n = lane & 15, mb = (lane >> 4) * 8;
  for (int r = 0; r < 8; ++r)
    dst[(size_t)(row0 + mb + r) * ld + col0 + n] = f2bf(c[r]);
}

// ---------------------------------------------------------------------------

__global__ void k_init_scores(float* scores, const float* b2, const float* ob2) {
  int t = blockIdx.x * blockDim.x + threadIdx.x;
  if (t < RR) scores[t] = b2[0] + ob2[0];
}

// cat1[r, 0:2048] = fc_target[b], cat1[r, 2048:4096] = fc_distr[b,i]  (bf16)
__global__ void k_cat1(const float* fct, const float* fcd, unsigned short* cat1) {
  int idx = blockIdx.x * blockDim.x + threadIdx.x;   // RR*TWOD
  int r = idx >> 12;
  int c = idx & 4095;
  int b = r >> 5;
  float v = (c < DD) ? fct[b * DD + c] : fcd[(size_t)r * DD + (c - DD)];
  cat1[idx] = f2bf(v);
}

// out[n*K + k] = bf16(in[k*Nc + n])  (K-contiguous storage for B fragments)
__global__ void k_transpose_bf(const float* in, unsigned short* out, int K, int Nc) {
  size_t idx = (size_t)blockIdx.x * blockDim.x + threadIdx.x;
  int k = (int)(idx / Nc);
  int n = (int)(idx % Nc);
  out[(size_t)n * K + k] = f2bf(in[idx]);
}

// att_target [B,1,36,D] f32 -> padded [B,48,D] bf16 (rows 36..47 = 0)
__global__ void k_pad_att_t(const float* att_t, unsigned short* out) {
  int idx = blockIdx.x * blockDim.x + threadIdx.x;   // BN*XP*DD
  int d = idx & (DD - 1);
  int t = idx >> 11;
  int x = t % XP;
  int b = t / XP;
  float v = (x < XX) ? att_t[((size_t)b * XX + x) * DD + d] : 0.0f;
  out[idx] = f2bf(v);
}

// tproj[b] = att_tp[b] (48x2048 bf16) @ WbilT (K-contig 2048x2048), stored bf16
// Register blocking: one wave owns all 3 x-tiles x 2 n-tiles (6 accumulators).
__global__ void k_gemm_tproj(const unsigned short* att_tp, const unsigned short* wbilT,
                             unsigned short* tproj) {
  int wid = (blockIdx.x * blockDim.x + threadIdx.x) >> 5;   // 16*64 waves
  int nt2 = wid & 63;
  int b   = wid >> 6;
  const unsigned short* A = att_tp + (size_t)b * XP * DD;
  int col0 = nt2 * 32;
  v8f c00 = {}, c01 = {}, c10 = {}, c11 = {}, c20 = {}, c21 = {};
  for (int k0 = 0; k0 < DD; k0 += 32) {
    v16bf a0 = frag_a(A, DD, 0,  k0);
    v16bf a1 = frag_a(A, DD, 16, k0);
    v16bf a2 = frag_a(A, DD, 32, k0);
    v16bf b0 = frag_b_kc(wbilT, DD, col0,      k0);
    v16bf b1 = frag_b_kc(wbilT, DD, col0 + 16, k0);
    c00 = wmma_bf16(a0, b0, c00);  c01 = wmma_bf16(a0, b1, c01);
    c10 = wmma_bf16(a1, b0, c10);  c11 = wmma_bf16(a1, b1, c11);
    c20 = wmma_bf16(a2, b0, c20);  c21 = wmma_bf16(a2, b1, c21);
  }
  unsigned short* Tp = tproj + (size_t)b * XP * DD;
  store_tile_bf16(Tp, DD, 0,  col0,      c00);
  store_tile_bf16(Tp, DD, 0,  col0 + 16, c01);
  store_tile_bf16(Tp, DD, 16, col0,      c10);
  store_tile_bf16(Tp, DD, 16, col0 + 16, c11);
  store_tile_bf16(Tp, DD, 32, col0,      c20);
  store_tile_bf16(Tp, DD, 32, col0 + 16, c21);
}

// inners[b,i] = tproj[b] (48x2048) @ att_d[b,i]^T, masked -> f32 [RR,48,48]
// One wave per (ri, yt): the f32 B panel is read & converted ONCE, reused for
// all 3 x-tiles (3 accumulators) -> single pass over the 151MB att_d stream.
__global__ void k_gemm_inners(const unsigned short* tproj, const float* att_d,
                              const int* mt, const int* md, float* inners) {
  int wid = (blockIdx.x * blockDim.x + threadIdx.x) >> 5;   // RR*3 waves
  int yt = wid % 3;
  int ri = wid / 3;
  int b  = ri >> 5;
  const unsigned short* A = tproj + (size_t)b * XP * DD;
  const float* Bsrc = att_d + (size_t)ri * XX * DD;    // [36 x 2048], K contiguous
  int lane = threadIdx.x & 31;
  v8f c0 = {}, c1 = {}, c2 = {};
  for (int k0 = 0; k0 < DD; k0 += 32) {
    if (k0 + 64 < DD) {           // prefetch next K panel of the big f32 stream
      int cprf = yt * 16 + (lane & 15);
      if (cprf < XX) __builtin_prefetch(Bsrc + (size_t)cprf * DD + k0 + 64, 0, 1);
    }
    v16bf bb = frag_b_f32(Bsrc, DD, yt * 16, k0, XX);
    v16bf a0 = frag_a(A, DD, 0,  k0);
    v16bf a1 = frag_a(A, DD, 16, k0);
    v16bf a2 = frag_a(A, DD, 32, k0);
    c0 = wmma_bf16(a0, bb, c0);
    c1 = wmma_bf16(a1, bb, c1);
    c2 = wmma_bf16(a2, bb, c2);
  }
  int n = lane & 15, mb = (lane >> 4) * 8;
  int y = yt * 16 + n;
  float* I = inners + (size_t)ri * XP * XP;
  bool y_ok = (y < XX) && (md[ri * XX + y] > 0);
  for (int r = 0; r < 8; ++r) {
    int x0 = mb + r;
    {
      bool ok = y_ok && (mt[b * XX + x0] > 0);           // x0 in 0..15 < 36
      I[x0 * XP + y] = ok ? c0[r] : BIGNEG;
    }
    {
      int x = 16 + x0;
      bool ok = y_ok && (mt[b * XX + x] > 0);            // x in 16..31 < 36
      I[x * XP + y] = ok ? c1[r] : BIGNEG;
    }
    {
      int x = 32 + x0;
      bool ok = (x < XX) && y_ok && (mt[b * XX + x] > 0);
      I[x * XP + y] = ok ? c2[r] : BIGNEG;
    }
  }
}

// per (b,i): tw = softmax_x(max_y inners), dw = softmax_y(max_x inners)
__global__ void k_soft(const float* inners, float* tw, float* dw) {
  int ri = blockIdx.x;
  const float* I = inners + (size_t)ri * XP * XP;
  __shared__ float sm[XX];
  __shared__ float Ms, Ss;
  int t = threadIdx.x;

  if (t < XX) {
    float m = -3.4e38f;
    for (int y = 0; y < XX; ++y) m = fmaxf(m, I[t * XP + y]);
    sm[t] = m;
  }
  __syncthreads();
  if (t == 0) {
    float M = sm[0];
    for (int x = 1; x < XX; ++x) M = fmaxf(M, sm[x]);
    float S = 0.f;
    for (int x = 0; x < XX; ++x) S += __expf(sm[x] - M);
    Ms = M; Ss = S;
  }
  __syncthreads();
  if (t < XX) tw[ri * XX + t] = __expf(sm[t] - Ms) / Ss;
  __syncthreads();

  if (t < XX) {
    float m = -3.4e38f;
    for (int x = 0; x < XX; ++x) m = fmaxf(m, I[x * XP + t]);
    sm[t] = m;
  }
  __syncthreads();
  if (t == 0) {
    float M = sm[0];
    for (int x = 1; x < XX; ++x) M = fmaxf(M, sm[x]);
    float S = 0.f;
    for (int x = 0; x < XX; ++x) S += __expf(sm[x] - M);
    Ms = M; Ss = S;
  }
  __syncthreads();
  if (t < XX) dw[ri * XX + t] = __expf(sm[t] - Ms) / Ss;
}

// cat2[r, 0:2048] = sum_x att_t[b,x,:]*tw[r,x]; cat2[r, 2048:] = sum_y att_d[r,y,:]*dw[r,y]
__global__ void k_feats(const float* att_t, const float* att_d,
                        const float* tw, const float* dw, unsigned short* cat2) {
  int idx = blockIdx.x * blockDim.x + threadIdx.x;   // RR*DD
  int d = idx & (DD - 1);
  int r = idx >> 11;
  int b = r >> 5;
  const float* At  = att_t + (size_t)b * XX * DD + d;
  const float* Ad  = att_d + (size_t)r * XX * DD + d;
  const float* twr = tw + r * XX;
  const float* dwr = dw + r * XX;
  float st = 0.f, sd = 0.f;
  for (int x = 0; x < XX; ++x) st += At[(size_t)x * DD] * twr[x];
  for (int y = 0; y < XX; ++y) sd += Ad[(size_t)y * DD] * dwr[y];
  cat2[(size_t)r * TWOD + d]      = f2bf(st);
  cat2[(size_t)r * TWOD + DD + d] = f2bf(sd);
}

// fused epilogue helper: scores[m] += sum_n relu(c[m,n] + bias[n]) * W2[n]
__device__ __forceinline__ void mlp_epilogue(v8f c, int row0, int col0,
                                             const float* bias1, const float* W2,
                                             float* scores) {
  int lane = threadIdx.x & 31;
  int n  = col0 + (lane & 15);
  int mb = (lane >> 4) * 8;
  float bia = bias1[n], w2 = W2[n];
  for (int r = 0; r < 8; ++r) {
    float v = c[r] + bia;
    v = v > 0.f ? v : 0.f;
    v *= w2;
    for (int off = 1; off < 16; off <<= 1) v += __shfl_xor(v, off, 32);
    if ((lane & 15) == 0) atomicAdd(&scores[row0 + mb + r], v);
  }
}

// scores += ( relu(A @ W + bias) ) @ W2   (2x2 register blocking)
__global__ void k_mlp(const unsigned short* A, const unsigned short* WT,
                      const float* bias1, const float* W2, float* scores) {
  int wid   = (blockIdx.x * blockDim.x + threadIdx.x) >> 5;  // 16*16 waves
  int nt2 = wid & 15;
  int mt2 = wid >> 4;
  int row0 = mt2 * 32, col0 = nt2 * 32;
  v8f c00 = {}, c01 = {}, c10 = {}, c11 = {};
  for (int k0 = 0; k0 < TWOD; k0 += 32) {
    v16bf a0 = frag_a(A, TWOD, row0,      k0);
    v16bf a1 = frag_a(A, TWOD, row0 + 16, k0);
    v16bf b0 = frag_b_kc(WT, TWOD, col0,      k0);
    v16bf b1 = frag_b_kc(WT, TWOD, col0 + 16, k0);
    c00 = wmma_bf16(a0, b0, c00);  c01 = wmma_bf16(a0, b1, c01);
    c10 = wmma_bf16(a1, b0, c10);  c11 = wmma_bf16(a1, b1, c11);
  }
  mlp_epilogue(c00, row0,      col0,      bias1, W2, scores);
  mlp_epilogue(c01, row0,      col0 + 16, bias1, W2, scores);
  mlp_epilogue(c10, row0 + 16, col0,      bias1, W2, scores);
  mlp_epilogue(c11, row0 + 16, col0 + 16, bias1, W2, scores);
}

// log_softmax over N=32 per batch (one wave32 per batch row)
__global__ void k_logsm(const float* scores, float* out) {
  int b = blockIdx.x, i = threadIdx.x;
  float v = scores[b * NN + i];
  float m = v;
  for (int off = 16; off >= 1; off >>= 1) m = fmaxf(m, __shfl_xor(m, off, 32));
  float e = __expf(v - m);
  float s = e;
  for (int off = 16; off >= 1; off >>= 1) s += __shfl_xor(s, off, 32);
  out[b * NN + i] = v - m - __logf(s);
}

// ---------------------------------------------------------------------------

extern "C" void kernel_launch(void* const* d_in, const int* in_sizes, int n_in,
                              void* d_out, int out_size, void* d_ws, size_t ws_size,
                              hipStream_t stream) {
  (void)in_sizes; (void)n_in; (void)out_size; (void)ws_size;
  const float* fct   = (const float*)d_in[0];   // [16,1,2048]
  const float* fcd   = (const float*)d_in[1];   // [16,32,2048]
  const float* att_t = (const float*)d_in[2];   // [16,1,36,2048]
  const float* att_d = (const float*)d_in[3];   // [16,32,36,2048]
  const int*   mt    = (const int*)d_in[4];     // [16,1,36]
  const int*   md    = (const int*)d_in[5];     // [16,32,36]
  const float* W1    = (const float*)d_in[6];   // [4096,512]
  const float* b1    = (const float*)d_in[7];
  const float* W2    = (const float*)d_in[8];   // [512,1]
  const float* b2    = (const float*)d_in[9];
  const float* Wbil  = (const float*)d_in[10];  // [2048,2048]
  const float* oW1   = (const float*)d_in[11];
  const float* ob1   = (const float*)d_in[12];
  const float* oW2   = (const float*)d_in[13];
  const float* ob2   = (const float*)d_in[14];
  float* out = (float*)d_out;

  // workspace carve-out (~36.5 MB), 256B aligned sub-buffers
  char* ws = (char*)d_ws;
  size_t off = 0;
  auto take = [&](size_t bytes) -> char* {
    char* p = ws + off;
    off = (off + bytes + 255) & ~(size_t)255;
    return p;
  };
  float*          scores = (float*)take((size_t)RR * 4);
  unsigned short* cat1   = (unsigned short*)take((size_t)RR * TWOD * 2);
  unsigned short* cat2   = (unsigned short*)take((size_t)RR * TWOD * 2);
  unsigned short* w1T    = (unsigned short*)take((size_t)TWOD * HH * 2);
  unsigned short* ow1T   = (unsigned short*)take((size_t)TWOD * HH * 2);
  unsigned short* wbT    = (unsigned short*)take((size_t)DD * DD * 2);
  unsigned short* attp   = (unsigned short*)take((size_t)BN * XP * DD * 2);
  unsigned short* tproj  = (unsigned short*)take((size_t)BN * XP * DD * 2);
  float*          inners = (float*)take((size_t)RR * XP * XP * 4);
  float*          tw     = (float*)take((size_t)RR * XX * 4);
  float*          dw     = (float*)take((size_t)RR * XX * 4);

  k_init_scores<<<2, 256, 0, stream>>>(scores, b2, ob2);
  k_cat1<<<(RR * TWOD) / 256, 256, 0, stream>>>(fct, fcd, cat1);
  k_transpose_bf<<<(TWOD * HH) / 256, 256, 0, stream>>>(W1, w1T, TWOD, HH);
  k_transpose_bf<<<(TWOD * HH) / 256, 256, 0, stream>>>(oW1, ow1T, TWOD, HH);
  k_transpose_bf<<<(DD * DD) / 256, 256, 0, stream>>>(Wbil, wbT, DD, DD);
  k_pad_att_t<<<(BN * XP * DD) / 256, 256, 0, stream>>>(att_t, attp);

  k_mlp<<<(16 * 16) / 8, 256, 0, stream>>>(cat1, w1T, b1, W2, scores);     // base scorer
  k_gemm_tproj<<<(BN * 64) / 8, 256, 0, stream>>>(attp, wbT, tproj);       // bilinear proj
  k_gemm_inners<<<(RR * 3) / 8, 256, 0, stream>>>(tproj, att_d, mt, md, inners);
  k_soft<<<RR, 64, 0, stream>>>(inners, tw, dw);
  k_feats<<<(RR * DD) / 256, 256, 0, stream>>>(att_t, att_d, tw, dw, cat2);
  k_mlp<<<(16 * 16) / 8, 256, 0, stream>>>(cat2, ow1T, ob1, oW2, scores);  // object scorer
  k_logsm<<<BN, 32, 0, stream>>>(scores, out);
}
